// Attention_60739427500161
// MI455X (gfx1250) — compile-verified, matrix operand order checked by reading, not run on hardware
//
#include <hip/hip_runtime.h>
#include <hip/hip_bf16.h>
#include <math.h>

// ---------------- problem constants ----------------
#define BATCH   2
#define S_LEN   2048
#define HIDDEN  2048
#define NH      32
#define NKV     8
#define HD      64
#define GQ      (NH / NKV)   // 4

typedef __bf16 bf16_t;
typedef __attribute__((ext_vector_type(16))) __bf16 v16bf;
typedef __attribute__((ext_vector_type(8)))  float  v8f;

union V16U { v16bf v; uint4 u[2]; };

// Load a 16x32 bf16 operand tile (A-layout == mirrored B-layout) from a
// row-major [16][stride] region. Lane m<16: row m, k=0..7 & 16..23;
// lane m+16: row m, k=8..15 & 24..31 -> two contiguous 16B loads per lane.
__device__ __forceinline__ v16bf load_tile16x32(const bf16_t* __restrict__ base,
                                                int stride, int lane) {
  int m    = lane & 15;
  int koff = (lane >> 4) << 3;       // 0 or 8
  const bf16_t* p = base + (size_t)m * stride + koff;
  V16U r;
  r.u[0] = *(const uint4*)(p);
  r.u[1] = *(const uint4*)(p + 16);
  return r.v;
}

__device__ __forceinline__ v8f wmma_bf16(v16bf a, v16bf b, v8f c) {
  return __builtin_amdgcn_wmma_f32_16x16x32_bf16(false, a, false, b, (short)0,
                                                 c, false, false);
}

// ---------------- elementwise converters ----------------
__global__ void convert_f32_bf16(const float* __restrict__ src,
                                 bf16_t* __restrict__ dst, int n) {
  int i = blockIdx.x * blockDim.x + threadIdx.x;
  if (i < n) dst[i] = (bf16_t)src[i];
}

// W [K][N] f32  ->  WT [N][K] bf16
__global__ void transpose_conv(const float* __restrict__ src,
                               bf16_t* __restrict__ dst, int K, int N) {
  int i = blockIdx.x * blockDim.x + threadIdx.x;
  if (i >= K * N) return;
  int k = i / N, n = i - k * N;
  dst[(size_t)n * K + k] = (bf16_t)src[i];
}

// ---------------- generic bf16 WMMA GEMM:  C[M][ldc] = A[M][K] * BT[N][K]^T
// block = 128 threads (4 waves); each wave computes a 32x64 f32 tile
// (8 accumulators, 2 A tiles reused over 4 B tiles -> 8 WMMA / 12 b128 loads).
__global__ void gemm_bf16_wmma(const bf16_t* __restrict__ A,
                               const bf16_t* __restrict__ BT,
                               float* __restrict__ C, int K, int ldc) {
  int lane = threadIdx.x & 31;
  int wave = threadIdx.x >> 5;
  int row0 = blockIdx.y * 128 + wave * 32;   // 32 rows per wave
  int col0 = blockIdx.x * 64;
  v8f acc00 = {}, acc01 = {}, acc02 = {}, acc03 = {};
  v8f acc10 = {}, acc11 = {}, acc12 = {}, acc13 = {};
  const bf16_t* arow0 = A + (size_t)row0 * K;
  const bf16_t* arow1 = A + (size_t)(row0 + 16) * K;
  const bf16_t* b0p = BT + (size_t)(col0 +  0) * K;
  const bf16_t* b1p = BT + (size_t)(col0 + 16) * K;
  const bf16_t* b2p = BT + (size_t)(col0 + 32) * K;
  const bf16_t* b3p = BT + (size_t)(col0 + 48) * K;
  for (int k0 = 0; k0 < K; k0 += 32) {
    v16bf a0 = load_tile16x32(arow0 + k0, K, lane);
    v16bf a1 = load_tile16x32(arow1 + k0, K, lane);
    v16bf b0 = load_tile16x32(b0p + k0, K, lane);
    v16bf b1 = load_tile16x32(b1p + k0, K, lane);
    v16bf b2 = load_tile16x32(b2p + k0, K, lane);
    v16bf b3 = load_tile16x32(b3p + k0, K, lane);
    acc00 = wmma_bf16(a0, b0, acc00);
    acc10 = wmma_bf16(a1, b0, acc10);
    acc01 = wmma_bf16(a0, b1, acc01);
    acc11 = wmma_bf16(a1, b1, acc11);
    acc02 = wmma_bf16(a0, b2, acc02);
    acc12 = wmma_bf16(a1, b2, acc12);
    acc03 = wmma_bf16(a0, b3, acc03);
    acc13 = wmma_bf16(a1, b3, acc13);
  }
  int rowl = 8 * (lane >> 4);
  int col  = lane & 15;
#pragma unroll
  for (int r = 0; r < 8; ++r) {
    size_t o0 = (size_t)(row0 + rowl + r) * ldc + col0 + col;
    C[o0]      = acc00[r];
    C[o0 + 16] = acc01[r];
    C[o0 + 32] = acc02[r];
    C[o0 + 48] = acc03[r];
    size_t o1 = (size_t)(row0 + 16 + rowl + r) * ldc + col0 + col;
    C[o1]      = acc10[r];
    C[o1 + 16] = acc11[r];
    C[o1 + 32] = acc12[r];
    C[o1 + 48] = acc13[r];
  }
}

// ---------------- RoPE: q part of qg -> bf16 Q[B][NH][S][D]
__global__ void rope_q_kernel(const float* __restrict__ qg,
                              const float* __restrict__ cosb,
                              const float* __restrict__ sinb,
                              bf16_t* __restrict__ qb) {
  int idx = blockIdx.x * blockDim.x + threadIdx.x;   // B*S*NH*HD threads
  int d = idx & (HD - 1);
  int t = idx >> 6;
  int h = t & (NH - 1);  t >>= 5;
  int s = t & (S_LEN - 1);
  int b = t >> 11;
  const float* row = qg + (size_t)(b * S_LEN + s) * (2 * HIDDEN);
  float x     = row[h * HD + d];
  float other = (d < HD / 2) ? -row[h * HD + d + HD / 2]
                             :  row[h * HD + d - HD / 2];
  float val = x * cosb[s * HD + d] + other * sinb[s * HD + d];
  qb[((size_t)(b * NH + h) * S_LEN + s) * HD + d] = (bf16_t)val;
}

// ---------------- RoPE: k proj -> bf16 K[B][NKV][S][D] and KT[B][NKV][D][S]
__global__ void rope_k_kernel(const float* __restrict__ kp,
                              const float* __restrict__ cosb,
                              const float* __restrict__ sinb,
                              bf16_t* __restrict__ kb,
                              bf16_t* __restrict__ kTb) {
  int idx = blockIdx.x * blockDim.x + threadIdx.x;   // B*S*NKV*HD threads
  int d  = idx & (HD - 1);
  int t  = idx >> 6;
  int kv = t & (NKV - 1);  t >>= 3;
  int s  = t & (S_LEN - 1);
  int b  = t >> 11;
  const float* row = kp + (size_t)(b * S_LEN + s) * (NKV * HD);
  float x     = row[kv * HD + d];
  float other = (d < HD / 2) ? -row[kv * HD + d + HD / 2]
                             :  row[kv * HD + d - HD / 2];
  float val = x * cosb[s * HD + d] + other * sinb[s * HD + d];
  bf16_t bv = (bf16_t)val;
  kb [((size_t)(b * NKV + kv) * S_LEN + s) * HD + d] = bv;
  kTb[((size_t)(b * NKV + kv) * HD + d) * S_LEN + s] = bv;
}

// ---------------- flash attention (causal, GQA, out = softmax(QK^T)·K),
// fused with sigmoid-gate and bf16 cast of the context.
// grid: (S/64, NH, B), block 128 (4 waves, one 16-query tile per wave).
__global__ void attn_gqa_wmma(const bf16_t* __restrict__ qb,
                              const bf16_t* __restrict__ kb,
                              const bf16_t* __restrict__ kTb,
                              const float* __restrict__ qg,   // gate = cols [HIDDEN,2*HIDDEN)
                              bf16_t* __restrict__ ctx) {
  __shared__ alignas(16) bf16_t pbuf[4][16][32];
  __shared__ float sstat[4][16];
  int lane = threadIdx.x & 31;
  int wave = threadIdx.x >> 5;
  int b  = blockIdx.z;
  int h  = blockIdx.y;
  int kv = h / GQ;
  int qbase = (blockIdx.x * 4 + wave) * 16;
  int half = lane >> 4;
  int col  = lane & 15;

  const bf16_t* Q  = qb  + ((size_t)(b * NH  + h ) * S_LEN + qbase) * HD;
  const bf16_t* Kb = kb  + (size_t)(b * NKV + kv) * S_LEN * HD;
  const bf16_t* KT = kTb + (size_t)(b * NKV + kv) * HD * S_LEN;

  v16bf a0 = load_tile16x32(Q,      HD, lane);   // Q[:,0:32]
  v16bf a1 = load_tile16x32(Q + 32, HD, lane);   // Q[:,32:64]

  v8f o0 = {}, o1 = {}, o2 = {}, o3 = {};        // O^T tiles: dims x queries
  float rmax[8], rsum[8];
#pragma unroll
  for (int r = 0; r < 8; ++r) { rmax[r] = -3.0e38f; rsum[r] = 0.f; }

  int nchunk = (qbase + 16 + 31) >> 5;           // 32-key chunks up to diagonal
  for (int j = 0; j < nchunk; ++j) {
    int kbase = j * 32;
    // scores: two 16x16 tiles, K-dim 64 -> 2 WMMAs each
    v8f s0 = {}, s1 = {};
    {
      v16bf bl = load_tile16x32(Kb + (size_t)kbase * HD,            HD, lane);
      v16bf bh = load_tile16x32(Kb + (size_t)kbase * HD + 32,       HD, lane);
      s0 = wmma_bf16(a0, bl, s0);  s0 = wmma_bf16(a1, bh, s0);
      v16bf cl = load_tile16x32(Kb + (size_t)(kbase + 16) * HD,     HD, lane);
      v16bf ch = load_tile16x32(Kb + (size_t)(kbase + 16) * HD + 32, HD, lane);
      s1 = wmma_bf16(a0, cl, s1);  s1 = wmma_bf16(a1, ch, s1);
    }
    int key0 = kbase + col;
    int key1 = kbase + 16 + col;
#pragma unroll
    for (int r = 0; r < 8; ++r) {
      int qrow = qbase + r + 8 * half;
      float v0 = s0[r] + ((key0 > qrow) ? -1e9f : 0.f);  // additive causal mask
      float v1 = s1[r] + ((key1 > qrow) ? -1e9f : 0.f);
      float mv = fmaxf(v0, v1);
#pragma unroll
      for (int off = 1; off < 16; off <<= 1) mv = fmaxf(mv, __shfl_xor(mv, off, 32));
      float nm = fmaxf(rmax[r], mv);
      float sc = __expf(rmax[r] - nm);
      rmax[r] = nm;
      float p0 = __expf(v0 - nm);
      float p1 = __expf(v1 - nm);
      float rs = p0 + p1;
#pragma unroll
      for (int off = 1; off < 16; off <<= 1) rs += __shfl_xor(rs, off, 32);
      rsum[r] = rsum[r] * sc + rs;
      pbuf[wave][r + 8 * half][col]      = (bf16_t)p0;   // P in [q][key] order
      pbuf[wave][r + 8 * half][col + 16] = (bf16_t)p1;
      if (col == 0) sstat[wave][r + 8 * half] = sc;
    }
    asm volatile("s_wait_dscnt 0" ::: "memory");   // per-wave LDS RAW fence
    float qscale = sstat[wave][col];               // scale per query column
    v16bf pB = load_tile16x32(&pbuf[wave][0][0], 32, lane);   // B-operand P^T
#pragma unroll
    for (int r = 0; r < 8; ++r) { o0[r] *= qscale; o1[r] *= qscale;
                                  o2[r] *= qscale; o3[r] *= qscale; }
    // O^T += K^T * P^T : A = K^T (dims x keys), contiguous from kTb
    v16bf ka0 = load_tile16x32(KT + (size_t) 0 * S_LEN + kbase, S_LEN, lane);
    v16bf ka1 = load_tile16x32(KT + (size_t)16 * S_LEN + kbase, S_LEN, lane);
    v16bf ka2 = load_tile16x32(KT + (size_t)32 * S_LEN + kbase, S_LEN, lane);
    v16bf ka3 = load_tile16x32(KT + (size_t)48 * S_LEN + kbase, S_LEN, lane);
    o0 = wmma_bf16(ka0, pB, o0);
    o1 = wmma_bf16(ka1, pB, o1);
    o2 = wmma_bf16(ka2, pB, o2);
    o3 = wmma_bf16(ka3, pB, o3);
  }

  // denom per query column via LDS (per-wave in-order DS)
  if (col == 0) {
#pragma unroll
    for (int r = 0; r < 8; ++r) sstat[wave][r + 8 * half] = rsum[r];
  }
  asm volatile("s_wait_dscnt 0" ::: "memory");
  float inv = 1.0f / sstat[wave][col];

  int qrow = qbase + col;
  const float* grow = qg + (size_t)(b * S_LEN + qrow) * (2 * HIDDEN) + HIDDEN + h * HD;
  bf16_t*      crow = ctx + (size_t)(b * S_LEN + qrow) * HIDDEN + h * HD;
#pragma unroll
  for (int r = 0; r < 8; ++r) {
    int d0 = r + 8 * half;
    float g, v;
    v = o0[r] * inv; g = grow[d0];      crow[d0]      = (bf16_t)(v / (1.f + __expf(-g)));
    v = o1[r] * inv; g = grow[d0 + 16]; crow[d0 + 16] = (bf16_t)(v / (1.f + __expf(-g)));
    v = o2[r] * inv; g = grow[d0 + 32]; crow[d0 + 32] = (bf16_t)(v / (1.f + __expf(-g)));
    v = o3[r] * inv; g = grow[d0 + 48]; crow[d0 + 48] = (bf16_t)(v / (1.f + __expf(-g)));
  }
}

// ---------------- host-side orchestration ----------------
extern "C" void kernel_launch(void* const* d_in, const int* in_sizes, int n_in,
                              void* d_out, int out_size, void* d_ws, size_t ws_size,
                              hipStream_t stream) {
  const float* hs   = (const float*)d_in[0];   // [B,S,HID]
  const float* cosb = (const float*)d_in[1];   // [S,D]
  const float* sinb = (const float*)d_in[2];   // [S,D]
  // d_in[3] = attention_mask (causal -1e9, applied analytically in-kernel)
  const float* wq = (const float*)d_in[4];     // [HID, 2*NH*D]
  const float* wk = (const float*)d_in[5];     // [HID, NKV*D]
  // d_in[6] = wv (dead in the reference computation)
  const float* wo = (const float*)d_in[7];     // [HID, HID]

  const int M = BATCH * S_LEN;                 // 4096 rows
  char* ws = (char*)d_ws;
  size_t off = 0;
  auto carve = [&](size_t bytes) { char* p = ws + off; off += (bytes + 255) & ~(size_t)255; return p; };

  bf16_t* hsb  = (bf16_t*)carve((size_t)M * HIDDEN * 2);            // bf16 hs
  bf16_t* wqT  = (bf16_t*)carve((size_t)(2 * HIDDEN) * HIDDEN * 2); // [4096][2048]
  bf16_t* wkT  = (bf16_t*)carve((size_t)(NKV * HD) * HIDDEN * 2);   // [512][2048]
  bf16_t* woT  = (bf16_t*)carve((size_t)HIDDEN * HIDDEN * 2);       // [2048][2048]
  float*  qgf  = (float*) carve((size_t)M * (2 * HIDDEN) * 4);      // q | gate
  float*  kpf  = (float*) carve((size_t)M * (NKV * HD) * 4);
  bf16_t* qbuf = (bf16_t*)carve((size_t)BATCH * NH  * S_LEN * HD * 2);
  bf16_t* kbuf = (bf16_t*)carve((size_t)BATCH * NKV * S_LEN * HD * 2);
  bf16_t* kTbf = (bf16_t*)carve((size_t)BATCH * NKV * HD * S_LEN * 2);
  bf16_t* ctx  = (bf16_t*)carve((size_t)M * HIDDEN * 2);
  (void)ws_size; (void)n_in; (void)in_sizes; (void)out_size;

  // 1) precision conversion / weight transposes
  {
    int n = M * HIDDEN;
    convert_f32_bf16<<<(n + 255) / 256, 256, 0, stream>>>(hs, hsb, n);
    int nq = HIDDEN * (2 * HIDDEN);
    transpose_conv<<<(nq + 255) / 256, 256, 0, stream>>>(wq, wqT, HIDDEN, 2 * HIDDEN);
    int nk = HIDDEN * (NKV * HD);
    transpose_conv<<<(nk + 255) / 256, 256, 0, stream>>>(wk, wkT, HIDDEN, NKV * HD);
    int no = HIDDEN * HIDDEN;
    transpose_conv<<<(no + 255) / 256, 256, 0, stream>>>(wo, woT, HIDDEN, HIDDEN);
  }

  // 2) projections: qg = hs@wq (N=4096), k = hs@wk (N=512)
  //    grid: x = N/64 col tiles, y = M/128 row tiles (32 rows per wave)
  gemm_bf16_wmma<<<dim3((2 * HIDDEN) / 64, M / 128), 128, 0, stream>>>(
      hsb, wqT, qgf, HIDDEN, 2 * HIDDEN);
  gemm_bf16_wmma<<<dim3((NKV * HD) / 64, M / 128), 128, 0, stream>>>(
      hsb, wkT, kpf, HIDDEN, NKV * HD);

  // 3) RoPE + layout for attention
  {
    int nq = BATCH * S_LEN * NH * HD;
    rope_q_kernel<<<nq / 256, 256, 0, stream>>>(qgf, cosb, sinb, qbuf);
    int nk = BATCH * S_LEN * NKV * HD;
    rope_k_kernel<<<nk / 256, 256, 0, stream>>>(kpf, cosb, sinb, kbuf, kTbf);
  }

  // 4) causal GQA flash attention (+ sigmoid(gate) fused)
  attn_gqa_wmma<<<dim3(S_LEN / 64, NH, BATCH), 128, 0, stream>>>(
      qbuf, kbuf, kTbf, qgf, ctx);

  // 5) output projection -> f32 result
  gemm_bf16_wmma<<<dim3(HIDDEN / 64, M / 128), 128, 0, stream>>>(
      ctx, woT, (float*)d_out, HIDDEN, HIDDEN);
}
